// LinearAttention2_9457517985870
// MI455X (gfx1250) — matrix operand order repaired and checked by reading, not compiled, not run on hardware
//
#include <hip/hip_runtime.h>
#include <hip/hip_bf16.h>
#include <math.h>

// Problem constants (match reference)
#define BB 8
#define NN 4096
#define CC 768
#define HH 12
#define DHD 64

typedef __attribute__((ext_vector_type(8)))  float  v8f;
typedef __attribute__((ext_vector_type(16))) __bf16 v16bf;
typedef unsigned int v4u  __attribute__((ext_vector_type(4)));
typedef int          v8i_ __attribute__((ext_vector_type(8)));
typedef int          v4i_ __attribute__((ext_vector_type(4)));

union Frag16 {
  uint4 u4[2];
  unsigned short us[16];
  v16bf bf;
};

__device__ __forceinline__ float bf2f(unsigned short u) {
  unsigned x = ((unsigned)u) << 16;
  float f; __builtin_memcpy(&f, &x, 4); return f;
}
__device__ __forceinline__ unsigned short f2bf(float f) {
  unsigned x; __builtin_memcpy(&x, &f, 4);
  x += 0x7FFFu + ((x >> 16) & 1u);           // round-to-nearest-even
  return (unsigned short)(x >> 16);
}
__device__ __forceinline__ v8f wmma_bf16(v16bf a, v16bf b, v8f c) {
  return __builtin_amdgcn_wmma_f32_16x16x32_bf16(false, a, false, b, (short)0, c,
                                                 false, false);
}

#if __has_builtin(__builtin_amdgcn_tensor_load_to_lds) && \
    __has_builtin(__builtin_amdgcn_s_wait_tensorcnt)
#define HAVE_TDM 1
#else
#define HAVE_TDM 0
#endif

#if HAVE_TDM
// Build a 2-D Tensor DMA descriptor (CDNA5 ISA) and issue
// TENSOR_LOAD_TO_LDS (6-arg clang-23 form). tile = rows x row_len elements
// (2B each), global rows separated by row_stride elements; stored contiguously
// at lds_off. Groups 2/3 zero (2-D tensor, tile_dim2=0).
__device__ __forceinline__ void tdm_load_2d(const void* gaddr, unsigned lds_off,
                                            unsigned rows, unsigned row_len,
                                            unsigned tensor_rows,
                                            unsigned row_stride_elems) {
  unsigned long long ga = (unsigned long long)(uintptr_t)gaddr;
  v4u g0;
  g0.x = 0x1u;                                    // count=1, user descriptor
  g0.y = lds_off;                                 // lds_addr (bytes)
  g0.z = (unsigned)(ga & 0xFFFFFFFFu);            // global_addr[31:0]
  g0.w = (unsigned)((ga >> 32) & 0x01FFFFFFu)     // global_addr[56:32]
         | 0x80000000u;                           // type=2 ("image") in [127:126]
  v8i_ g1;
  g1[0] = (int)(1u << 16);                                      // data_size=1 -> 2B
  g1[1] = (int)((row_len & 0xFFFFu) << 16);                     // tensor_dim0 lo
  g1[2] = (int)(((tensor_rows & 0xFFFFu) << 16) |               // tensor_dim1 lo
                ((row_len >> 16) & 0xFFFFu));                   // tensor_dim0 hi
  g1[3] = (int)(((row_len & 0xFFFFu) << 16) |                   // tile_dim0
                ((tensor_rows >> 16) & 0xFFFFu));               // tensor_dim1 hi
  g1[4] = (int)(rows & 0xFFFFu);                                // tile_dim1 (dim2=0)
  g1[5] = (int)row_stride_elems;                                // dim0_stride lo
  g1[6] = 0;                                                    // dim0_stride hi
  g1[7] = 0;
  v4i_ z4 = {0, 0, 0, 0};
  v8i_ z8 = {0, 0, 0, 0, 0, 0, 0, 0};
  __builtin_amdgcn_tensor_load_to_lds(g0, g1, z4, z4, z8, 0);
}
#endif

// ---------------------------------------------------------------------------
// fp32 -> bf16 bulk convert (vec4)
// ---------------------------------------------------------------------------
__global__ __launch_bounds__(256)
void cvt_bf16_kernel(const float* __restrict__ src, unsigned short* __restrict__ dst,
                     int count4) {
  int i = blockIdx.x * 256 + threadIdx.x;
  if (i >= count4) return;
  float4 v = ((const float4*)src)[i];
  ((ushort4*)dst)[i] = make_ushort4(f2bf(v.x), f2bf(v.y), f2bf(v.z), f2bf(v.w));
}

// ---------------------------------------------------------------------------
// GEMM: Out[M,N] = A[M,K] @ W[N,K]^T   (A,W bf16; acc f32)
//   - per wave: 32(M) x 64(N) macro tile, K-step 32, 8 WMMAs/step
//   - epilogue: phi(v)=elu(v)+1 for columns < phi_limit (QKV: q,k regions)
//   - OUT_F32: f32 output + bias (proj); else bf16 output
// ---------------------------------------------------------------------------
template <bool OUT_F32>
__global__ __launch_bounds__(256)
void gemm_bf16_kernel(const unsigned short* __restrict__ A,
                      const unsigned short* __restrict__ W,
                      void* __restrict__ Out,
                      const float* __restrict__ bias,
                      int M, int N, int K, int phi_limit) {
  const int lane = threadIdx.x & 31;
  const int widx = threadIdx.x >> 5;
  const int wave = blockIdx.x * 8 + widx;
  const int Mt   = M >> 5;
  const int mt   = wave % Mt;
  const int nt   = wave / Mt;
  const int m0   = mt << 5;
  const int n0   = nt << 6;
  const int lh   = lane >> 4;
  const int li   = lane & 15;

  v8f acc[2][4];
  const v8f vzero = {};
  for (int i = 0; i < 2; ++i)
    for (int j = 0; j < 4; ++j) acc[i][j] = vzero;

  const unsigned short* arow0 = A + (size_t)(m0 + li) * K;
  const unsigned short* arow1 = A + (size_t)(m0 + 16 + li) * K;

  for (int k0 = 0; k0 < K; k0 += 32) {
    Frag16 af[2];
    {
      const unsigned short* p0 = arow0 + k0 + lh * 8;   // K chunk0 / chunk0+16
      af[0].u4[0] = *(const uint4*)(p0);
      af[0].u4[1] = *(const uint4*)(p0 + 16);
      const unsigned short* p1 = arow1 + k0 + lh * 8;
      af[1].u4[0] = *(const uint4*)(p1);
      af[1].u4[1] = *(const uint4*)(p1 + 16);
    }
    Frag16 bfrag[4];
#pragma unroll
    for (int j = 0; j < 4; ++j) {
      const unsigned short* wp = W + (size_t)(n0 + j * 16 + li) * K + k0 + lh * 8;
      bfrag[j].u4[0] = *(const uint4*)(wp);
      bfrag[j].u4[1] = *(const uint4*)(wp + 16);
    }
    if (k0 + 32 < K) {                                   // gfx1250 global_prefetch_b8
      __builtin_prefetch(arow0 + k0 + 32, 0, 1);
      __builtin_prefetch(W + (size_t)(n0 + li) * K + k0 + 32, 0, 1);
    }
#pragma unroll
    for (int i = 0; i < 2; ++i)
#pragma unroll
      for (int j = 0; j < 4; ++j)
        acc[i][j] = wmma_bf16(af[i].bf, bfrag[j].bf, acc[i][j]);
  }

  // Epilogue. C layout: VGPR r -> row m0 + i*16 + lh*8 + r, col n0 + j*16 + li
#pragma unroll
  for (int j = 0; j < 4; ++j) {
    const int col     = n0 + j * 16 + li;
    const bool do_phi = col < phi_limit;
    const float bv    = OUT_F32 ? bias[col] : 0.0f;
#pragma unroll
    for (int i = 0; i < 2; ++i) {
      const int rbase = m0 + i * 16 + lh * 8;
#pragma unroll
      for (int r = 0; r < 8; ++r) {
        float v = acc[i][j][r];
        if (do_phi) v = (v > 0.0f) ? (v + 1.0f) : __expf(v);  // elu(v)+1
        const size_t idx = (size_t)(rbase + r) * N + col;
        if (OUT_F32) ((float*)Out)[idx] = v + bv;
        else         ((unsigned short*)Out)[idx] = f2bf(v);
      }
    }
  }
}

// ---------------------------------------------------------------------------
// ksum[b,h,d] = sum_n phi(k)[b,h,n,d]
// ---------------------------------------------------------------------------
__global__ __launch_bounds__(256)
void ksum_kernel(const unsigned short* __restrict__ qkv_b, float* __restrict__ ksum) {
  int idx = blockIdx.x * 256 + threadIdx.x;
  if (idx >= BB * HH * DHD) return;
  int d  = idx & 63;
  int bh = idx >> 6;
  int b  = bh / HH, h = bh % HH;
  const unsigned short* kp = qkv_b + ((size_t)b * NN) * (3 * CC) + CC + h * DHD + d;
  float s = 0.0f;
  for (int n = 0; n < NN; ++n) s += bf2f(kp[(size_t)n * (3 * CC)]);
  ksum[idx] = s;
}

// ---------------------------------------------------------------------------
// kv[b,h,d,e] = sum_n phi(k)[n,d] * v[n,e]   (M=N=64, K=4096)
// One block per (b,h); 16 waves, one 16x16 tile each.
// TDM stages 128(n) x 64 tiles of phi(K) and V into LDS (strided global rows ->
// contiguous LDS), waves gather K-major fragments from LDS via ds loads.
// ---------------------------------------------------------------------------
#define KV_CHUNK 128
__global__ __launch_bounds__(512)
void kv_kernel(const unsigned short* __restrict__ qkv_b, float* __restrict__ kv_out) {
  const int bh   = blockIdx.x;
  const int b    = bh / HH, h = bh % HH;
  const int lane = threadIdx.x & 31, w = threadIdx.x >> 5;
  const int d0   = (w & 3) * 16, e0 = (w >> 2) * 16;
  const int lh   = lane >> 4, li = lane & 15;
  const unsigned short* kbase = qkv_b + ((size_t)b * NN) * (3 * CC) + CC + h * DHD;
  const unsigned short* vbase = qkv_b + ((size_t)b * NN) * (3 * CC) + 2 * CC + h * DHD;
  const int d = d0 + li, e = e0 + li;
  v8f acc = {};

#if HAVE_TDM
  __shared__ unsigned short smem[2 * KV_CHUNK * DHD];   // 32 KB: [K tile][V tile]
  unsigned short* sk = smem;
  unsigned short* sv = smem + KV_CHUNK * DHD;
  const unsigned lds_k = (unsigned)(uintptr_t)sk;       // LDS byte offset
  const unsigned lds_v = (unsigned)(uintptr_t)sv;

  for (int c0 = 0; c0 < NN; c0 += KV_CHUNK) {
    __syncthreads();                      // previous chunk fully consumed
    if (threadIdx.x == 0) {               // TDM ignores EXEC; one issue per block
      tdm_load_2d(kbase + (size_t)c0 * (3 * CC), lds_k, KV_CHUNK, DHD, NN, 3 * CC);
      tdm_load_2d(vbase + (size_t)c0 * (3 * CC), lds_v, KV_CHUNK, DHD, NN, 3 * CC);
      __builtin_amdgcn_s_wait_tensorcnt(0);
    }
    __syncthreads();                      // tiles visible to all waves
#pragma unroll
    for (int s = 0; s < KV_CHUNK / 32; ++s) {
      const int nn = s * 32;
      Frag16 a, bb;
#pragma unroll
      for (int j = 0; j < 8; ++j) {       // A = phi(K)^T column gather from LDS
        a.us[j]     = sk[(nn + lh * 8 + j) * DHD + d];
        a.us[j + 8] = sk[(nn + 16 + lh * 8 + j) * DHD + d];
      }
#pragma unroll
      for (int j = 0; j < 16; ++j)        // B = V column gather from LDS
        bb.us[j] = sv[(nn + (j >> 3) * 16 + lh * 8 + (j & 7)) * DHD + e];
      acc = wmma_bf16(a.bf, bb.bf, acc);
    }
  }
#else
  for (int n0 = 0; n0 < NN; n0 += 32) {
    Frag16 a, bb;
#pragma unroll
    for (int j = 0; j < 8; ++j) {
      a.us[j]     = kbase[(size_t)(n0 + lh * 8 + j) * (3 * CC) + d];
      a.us[j + 8] = kbase[(size_t)(n0 + 16 + lh * 8 + j) * (3 * CC) + d];
    }
#pragma unroll
    for (int j = 0; j < 16; ++j)
      bb.us[j] = vbase[(size_t)(n0 + (j >> 3) * 16 + lh * 8 + (j & 7)) * (3 * CC) + e];
    acc = wmma_bf16(a.bf, bb.bf, acc);
  }
#endif

  float* outp = kv_out + (size_t)bh * (DHD * DHD);
#pragma unroll
  for (int r = 0; r < 8; ++r)
    outp[(size_t)(d0 + lh * 8 + r) * DHD + e] = acc[r];
}

// ---------------------------------------------------------------------------
// z[b,h,n] = 1 / (dot(phi(q)[n,:], ksum[b,h,:]) + eps)
// ---------------------------------------------------------------------------
__global__ __launch_bounds__(256)
void z_kernel(const unsigned short* __restrict__ qkv_b, const float* __restrict__ ksum,
              float* __restrict__ zbuf) {
  int idx = blockIdx.x * 256 + threadIdx.x;   // exactly B*H*N threads launched
  int n  = idx & (NN - 1);
  int bh = idx >> 12;
  int b  = bh / HH, h = bh % HH;
  const unsigned short* qp = qkv_b + ((size_t)(b * NN + n)) * (3 * CC) + h * DHD;
  const float* ks = ksum + bh * DHD;
  float dot = 0.0f;
#pragma unroll 8
  for (int d = 0; d < DHD; ++d) dot += bf2f(qp[d]) * ks[d];
  zbuf[idx] = 1.0f / (dot + 1e-6f);
}

// ---------------------------------------------------------------------------
// attn[b,n,h*64+e] = z[b,h,n] * sum_d phi(q)[n,d]*kv[d,e]   (K=64, 2 steps)
// Per wave: 32(n) x 64(e) tile for one (b,h). bf16 output for proj GEMM.
// ---------------------------------------------------------------------------
__global__ __launch_bounds__(256)
void apply_kernel(const unsigned short* __restrict__ qkv_b,
                  const float* __restrict__ kv,
                  const float* __restrict__ zbuf,
                  unsigned short* __restrict__ attn_b) {
  const int lane = threadIdx.x & 31, widx = threadIdx.x >> 5;
  const int wave = blockIdx.x * 8 + widx;
  const int bh   = wave >> 7;           // 128 m-tiles per head
  const int mt   = wave & 127;
  const int b    = bh / HH, h = bh % HH;
  const int n0   = mt << 5;
  const int lh   = lane >> 4, li = lane & 15;

  const float* kvp = kv + (size_t)bh * (DHD * DHD);
  Frag16 bfrag[2][4];                   // kv columns, f32 -> bf16, loaded once
  for (int kk = 0; kk < 2; ++kk)
    for (int j = 0; j < 4; ++j) {
      const int e = j * 16 + li;
#pragma unroll
      for (int t = 0; t < 16; ++t)
        bfrag[kk][j].us[t] = f2bf(kvp[(size_t)(kk * 32 + (t >> 3) * 16 + lh * 8 + (t & 7)) * DHD + e]);
    }

  v8f acc[2][4];
  const v8f vzero = {};
  for (int i = 0; i < 2; ++i)
    for (int j = 0; j < 4; ++j) acc[i][j] = vzero;

  const unsigned short* qbase = qkv_b + ((size_t)b * NN) * (3 * CC) + h * DHD;
  for (int kk = 0; kk < 2; ++kk) {
    Frag16 af[2];
#pragma unroll
    for (int i = 0; i < 2; ++i) {
      const unsigned short* rp =
          qbase + (size_t)(n0 + i * 16 + li) * (3 * CC) + kk * 32 + lh * 8;
      af[i].u4[0] = *(const uint4*)rp;
      af[i].u4[1] = *(const uint4*)(rp + 16);
    }
#pragma unroll
    for (int i = 0; i < 2; ++i)
#pragma unroll
      for (int j = 0; j < 4; ++j)
        acc[i][j] = wmma_bf16(af[i].bf, bfrag[kk][j].bf, acc[i][j]);
  }

  const float* zp = zbuf + (size_t)bh * NN;
#pragma unroll
  for (int i = 0; i < 2; ++i) {
    const int rb = n0 + i * 16 + lh * 8;
    float zr[8];
#pragma unroll
    for (int r = 0; r < 8; ++r) zr[r] = zp[rb + r];
#pragma unroll
    for (int j = 0; j < 4; ++j) {
      const int e = j * 16 + li;
#pragma unroll
      for (int r = 0; r < 8; ++r)
        attn_b[(size_t)(b * NN + rb + r) * CC + h * DHD + e] = f2bf(acc[i][j][r] * zr[r]);
    }
  }
}

// ---------------------------------------------------------------------------
extern "C" void kernel_launch(void* const* d_in, const int* in_sizes, int n_in,
                              void* d_out, int out_size, void* d_ws, size_t ws_size,
                              hipStream_t stream) {
  const float* x      = (const float*)d_in[0];
  const float* qkv_w  = (const float*)d_in[1];
  const float* proj_w = (const float*)d_in[2];
  const float* proj_b = (const float*)d_in[3];

  char* ws = (char*)d_ws;
  unsigned short* x_b     = (unsigned short*)(ws + 0);            // 48 MB
  unsigned short* qkvw_b  = (unsigned short*)(ws + 50331648ull);  // 3.375 MB
  unsigned short* projw_b = (unsigned short*)(ws + 53870592ull);  // 1.125 MB
  unsigned short* qkv_b   = (unsigned short*)(ws + 55050240ull);  // 144 MB
  unsigned short* attn_b  = (unsigned short*)(ws + 206045184ull); // 48 MB
  float*          ksum    = (float*)(ws + 256376832ull);          // 24 KB
  float*          zbuf    = (float*)(ws + 256401408ull);          // 1.5 MB

  float* out_f = (float*)d_out;                      // [B,N,C] f32
  float* kv    = out_f + (size_t)BB * NN * CC;       // [B,H,64,64] f32 (2nd output)

  // 1) bf16 packing
  cvt_bf16_kernel<<<24576, 256, 0, stream>>>(x, x_b, (BB * NN * CC) / 4);
  cvt_bf16_kernel<<<1728, 256, 0, stream>>>(qkv_w, qkvw_b, (3 * CC * CC) / 4);
  cvt_bf16_kernel<<<576, 256, 0, stream>>>(proj_w, projw_b, (CC * CC) / 4);

  // 2) QKV GEMM, phi fused on q,k columns (< 2C); bf16 out
  gemm_bf16_kernel<false><<<4608, 256, 0, stream>>>(x_b, qkvw_b, (void*)qkv_b, nullptr,
                                                    BB * NN, 3 * CC, CC, 2 * CC);
  // 3) ksum, 4) kv state via TDM+WMMA (direct to d_out), 5) z
  ksum_kernel<<<24, 256, 0, stream>>>(qkv_b, ksum);
  kv_kernel<<<BB * HH, 512, 0, stream>>>(qkv_b, kv);
  z_kernel<<<1536, 256, 0, stream>>>(qkv_b, ksum, zbuf);
  // 6) out_head = z * (phi(q) @ kv) -> bf16 attn
  apply_kernel<<<1536, 256, 0, stream>>>(qkv_b, kv, zbuf, attn_b);
  // 7) proj GEMM + bias -> f32 d_out
  gemm_bf16_kernel<true><<<1536, 256, 0, stream>>>(attn_b, projw_b, d_out, proj_b,
                                                   BB * NN, CC, CC, 0);
}
